// SmallThinkerMoeBlock_79121887527466
// MI455X (gfx1250) — matrix operand order repaired and checked by reading, not compiled, not run on hardware
//
#include <hip/hip_runtime.h>
#include <hip/hip_bf16.h>

// Problem sizes (compile-time, match reference)
constexpr int T = 2048;   // tokens
constexpr int H = 1024;   // hidden
constexpr int F = 512;    // ffn hidden
constexpr int E = 32;     // experts
constexpr int K = 4;      // top-k
constexpr int MT = 32;    // token tile per expert block (2 WMMA M-tiles)
constexpr int HPAD = 8;   // LDS row padding (bf16 elems) to break bank conflicts

typedef __attribute__((ext_vector_type(16))) __bf16 v16bf;
typedef __attribute__((ext_vector_type(4)))  __bf16 v4bf;
typedef __attribute__((ext_vector_type(8)))  float  v8f;

__device__ inline v8f vzero8() {
  v8f z;
#pragma unroll
  for (int i = 0; i < 8; ++i) z[i] = 0.f;
  return z;
}

// ---- fragment loaders (overloaded on operand storage type) -----------------
// A-fragment: frag[0..7]=p0[0..7], frag[8..15]=p1[0..7] (two K-octets)
__device__ inline v16bf fragA88(const float* __restrict__ p0,
                                const float* __restrict__ p1) {
  v16bf r;
#pragma unroll
  for (int i = 0; i < 8; ++i) { r[i] = (__bf16)p0[i]; r[8 + i] = (__bf16)p1[i]; }
  return r;
}
__device__ inline v16bf fragA88(const __bf16* __restrict__ p0,
                                const __bf16* __restrict__ p1) {
  v16bf r;
#pragma unroll
  for (int i = 0; i < 8; ++i) { r[i] = p0[i]; r[8 + i] = p1[i]; }
  return r;
}
// B-fragment: 16 consecutive K values
__device__ inline v16bf fragB16(const float* __restrict__ p) {
  v16bf r;
#pragma unroll
  for (int i = 0; i < 16; ++i) r[i] = (__bf16)p[i];
  return r;
}
__device__ inline v16bf fragB16(const __bf16* __restrict__ p) {
  v16bf r;
#pragma unroll
  for (int i = 0; i < 16; ++i) r[i] = p[i];
  return r;
}

__device__ inline v8f wmma_bf16(v16bf a, v16bf b, v8f c) {
  return __builtin_amdgcn_wmma_f32_16x16x32_bf16(false, a, false, b,
                                                 (short)0, c, false, false);
}

// ---------------------------------------------------------------------------
// fp32 -> bf16 bulk conversion (grid-stride, float4 in / 4x bf16 out)
// ---------------------------------------------------------------------------
__global__ __launch_bounds__(256) void cvt_bf16_kernel(
    const float* __restrict__ src, __bf16* __restrict__ dst, long long n) {
  const long long stride = (long long)gridDim.x * blockDim.x * 4;
  for (long long i = ((long long)blockIdx.x * blockDim.x + threadIdx.x) * 4;
       i < n; i += stride) {
    const float4 v = *(const float4*)(src + i);
    v4bf o;
    o[0] = (__bf16)v.x; o[1] = (__bf16)v.y; o[2] = (__bf16)v.z; o[3] = (__bf16)v.w;
    *(v4bf*)(dst + i) = o;
  }
}

// ---------------------------------------------------------------------------
// Router: one wave per token; lane l owns expert l. Computes logits, top-4
// (+softmax), writes logits to d_out tail, appends (token, weight) per expert.
// ---------------------------------------------------------------------------
__global__ __launch_bounds__(256) void router_kernel(
    const float* __restrict__ xr, const float* __restrict__ wr,
    float* __restrict__ logits_out, int* __restrict__ tlist,
    float* __restrict__ wlist, int* __restrict__ counts) {
  const int wave = threadIdx.x >> 5;
  const int lane = threadIdx.x & 31;
  const int t = blockIdx.x * 8 + wave;
  if (t >= T) return;

  const float4* x = (const float4*)(xr + (size_t)t * H);
  const float4* w = (const float4*)(wr + (size_t)lane * H);
  float acc = 0.f;
#pragma unroll 4
  for (int i = 0; i < H / 4; ++i) {
    float4 a = x[i], b = w[i];
    acc += a.x * b.x + a.y * b.y + a.z * b.z + a.w * b.w;
  }
  logits_out[t * E + lane] = acc;

  // Orderable key; low bits = inverted index so ties pick lowest expert.
  unsigned u = __float_as_uint(acc);
  u = (u & 0x80000000u) ? ~u : (u | 0x80000000u);
  unsigned long long key =
      ((unsigned long long)u << 32) | (unsigned)(E - 1 - lane);

  float sel_logit[K];
  int sel_idx[K];
  unsigned long long mykey = key;
#pragma unroll
  for (int kk = 0; kk < K; ++kk) {
    unsigned long long m = mykey;
#pragma unroll
    for (int off = 16; off >= 1; off >>= 1) {
      unsigned long long o = __shfl_xor(m, off);
      m = (o > m) ? o : m;
    }
    const int widx = E - 1 - (int)(m & 0xFFu);
    sel_idx[kk] = widx;
    unsigned lu = (unsigned)(m >> 32);
    lu = (lu & 0x80000000u) ? (lu & 0x7FFFFFFFu) : ~lu;
    sel_logit[kk] = __uint_as_float(lu);
    if (lane == widx) mykey = 0ull;
  }

  const float mx = sel_logit[0];
  float ex[K], s = 0.f;
#pragma unroll
  for (int kk = 0; kk < K; ++kk) { ex[kk] = __expf(sel_logit[kk] - mx); s += ex[kk]; }
  const float inv = 1.f / s;

  if (lane < K) {
    const int e = sel_idx[lane];
    const int pos = atomicAdd(&counts[e], 1);
    tlist[e * T + pos] = t;
    wlist[e * T + pos] = ex[lane] * inv;
  }
}

// ---------------------------------------------------------------------------
// Fused expert kernel: block = (expert e, tile of 32 gathered tokens).
// Each wave handles one 64/128-col N strip and BOTH 16-row M tiles, so every
// B fragment is reused twice from registers (halves weight traffic).
// GEMM1: h[32,F] = relu(X Wg^T) * (X Wu^T) * weight -> LDS (bf16)
// GEMM2: out[32,H] += h @ Wd^T (atomic f32 scatter into d_out)
// ---------------------------------------------------------------------------
template <typename XT, typename WT>
__global__ __launch_bounds__(256) void expert_kernel(
    const XT* __restrict__ X, const WT* __restrict__ Wg,
    const WT* __restrict__ Wu, const WT* __restrict__ Wd,
    const int* __restrict__ tlist, const float* __restrict__ wlist,
    const int* __restrict__ counts, float* __restrict__ out) {
  const int e = blockIdx.x >> 6;     // 32 experts
  const int tile = blockIdx.x & 63;  // up to T/MT tiles
  const int cnt = counts[e];
  if (tile * MT >= cnt) return;      // uniform across block

  __shared__ int toks[MT];
  __shared__ float wts[MT];
  __shared__ __bf16 hbuf[MT][F + HPAD];

  const int tid = threadIdx.x;
  if (tid < MT) {
    const int i = tile * MT + tid;
    const bool v = i < cnt;
    toks[tid] = v ? tlist[e * T + i] : tlist[e * T];  // cnt>=1 here
    wts[tid] = v ? wlist[e * T + i] : 0.f;            // zero-weight padding
  }
  __syncthreads();

  const int wave = tid >> 5;
  const int lane = tid & 31;
  const int nrow = lane & 15;             // N (and A-row M) index
  const int khalf = (lane < 16) ? 0 : 1;  // K-octet half selector
  const int koffA = khalf * 8;
  const int koffB = khalf * 16;

  const WT* wg = Wg + (size_t)e * F * H;
  const WT* wu = Wu + (size_t)e * F * H;
  const WT* wd = Wd + (size_t)e * H * F;

  // ---------------- GEMM1 ----------------
  {
    v8f accG[2][4], accU[2][4];
#pragma unroll
    for (int m = 0; m < 2; ++m)
#pragma unroll
      for (int i = 0; i < 4; ++i) { accG[m][i] = vzero8(); accU[m][i] = vzero8(); }

    const XT* xrow0 = X + (size_t)toks[nrow] * H;
    const XT* xrow1 = X + (size_t)toks[16 + nrow] * H;
    const int nb = wave * 64;  // 64 F-columns per wave

    for (int k0 = 0; k0 < H; k0 += 32) {
      const v16bf a0 = fragA88(xrow0 + k0 + koffB, xrow0 + k0 + koffB + 8);
      const v16bf a1 = fragA88(xrow1 + k0 + koffB, xrow1 + k0 + koffB + 8);
#pragma unroll
      for (int nt = 0; nt < 4; ++nt) {
        const int f = nb + nt * 16 + nrow;
        const v16bf bg = fragB16(wg + (size_t)f * H + k0 + koffB);
        accG[0][nt] = wmma_bf16(a0, bg, accG[0][nt]);
        accG[1][nt] = wmma_bf16(a1, bg, accG[1][nt]);
        const v16bf bu = fragB16(wu + (size_t)f * H + k0 + koffB);
        accU[0][nt] = wmma_bf16(a0, bu, accU[0][nt]);
        accU[1][nt] = wmma_bf16(a1, bu, accU[1][nt]);
      }
    }
    // C/D layout: VGPR r, lanes<16 -> M=r, lanes>=16 -> M=r+8; N = lane&15
#pragma unroll
    for (int mt = 0; mt < 2; ++mt) {
#pragma unroll
      for (int nt = 0; nt < 4; ++nt) {
#pragma unroll
        for (int r = 0; r < 8; ++r) {
          const int m = mt * 16 + r + khalf * 8;
          const float g = accG[mt][nt][r];
          const float hv = (g > 0.f ? g : 0.f) * accU[mt][nt][r] * wts[m];
          hbuf[m][nb + nt * 16 + nrow] = (__bf16)hv;
        }
      }
    }
  }
  __syncthreads();

  // ---------------- GEMM2 ----------------
  {
    v8f acc[2][8];
#pragma unroll
    for (int m = 0; m < 2; ++m)
#pragma unroll
      for (int i = 0; i < 8; ++i) acc[m][i] = vzero8();

    const __bf16* h0 = &hbuf[nrow][0];
    const __bf16* h1 = &hbuf[16 + nrow][0];
    const int nb = wave * 128;  // 128 H-columns per wave

    for (int k0 = 0; k0 < F; k0 += 32) {
      const v16bf a0 = fragA88(h0 + k0 + koffB, h0 + k0 + koffB + 8);
      const v16bf a1 = fragA88(h1 + k0 + koffB, h1 + k0 + koffB + 8);
#pragma unroll
      for (int nt = 0; nt < 8; ++nt) {
        const int n = nb + nt * 16 + nrow;
        const v16bf b = fragB16(wd + (size_t)n * F + k0 + koffB);
        acc[0][nt] = wmma_bf16(a0, b, acc[0][nt]);
        acc[1][nt] = wmma_bf16(a1, b, acc[1][nt]);
      }
    }
#pragma unroll
    for (int mt = 0; mt < 2; ++mt) {
#pragma unroll
      for (int r = 0; r < 8; ++r) {
        const int m = mt * 16 + r + khalf * 8;
        if (wts[m] != 0.f) {  // skip zero-weight padding rows
          float* orow = out + (size_t)toks[m] * H;
#pragma unroll
          for (int nt = 0; nt < 8; ++nt) {
            atomicAdd(orow + nb + nt * 16 + nrow, acc[mt][nt][r]);
          }
        }
      }
    }
  }
}

// ---------------------------------------------------------------------------
extern "C" void kernel_launch(void* const* d_in, const int* in_sizes, int n_in,
                              void* d_out, int out_size, void* d_ws,
                              size_t ws_size, hipStream_t stream) {
  const float* router_input = (const float*)d_in[0];  // [T,H]
  const float* hidden = (const float*)d_in[1];        // [T,H]
  const float* w_router = (const float*)d_in[2];      // [E,H]
  const float* w_gate = (const float*)d_in[3];        // [E,F,H]
  const float* w_up = (const float*)d_in[4];          // [E,F,H]
  const float* w_down = (const float*)d_in[5];        // [E,H,F]

  float* out = (float*)d_out;           // [T,H]
  float* logits = out + (size_t)T * H;  // [T,E]

  char* ws = (char*)d_ws;
  size_t off = 0;
  int* counts = (int*)(ws + off);  off += 256;
  int* tlist = (int*)(ws + off);   off += (size_t)E * T * sizeof(int);
  float* wlist = (float*)(ws + off); off += (size_t)E * T * sizeof(float);
  off = (off + 255) & ~(size_t)255;

  const long long nX = (long long)T * H;
  const long long nW = (long long)E * F * H;  // same count for gate/up/down
  __bf16* Xbf = (__bf16*)(ws + off);  off += (size_t)nX * 2;
  __bf16* Wgbf = (__bf16*)(ws + off); off += (size_t)nW * 2;
  __bf16* Wubf = (__bf16*)(ws + off); off += (size_t)nW * 2;
  __bf16* Wdbf = (__bf16*)(ws + off); off += (size_t)nW * 2;
  const bool bf16_path = ws_size >= off;  // deterministic per harness

  hipMemsetAsync(counts, 0, E * sizeof(int), stream);
  hipMemsetAsync(out, 0, (size_t)T * H * sizeof(float), stream);

  router_kernel<<<T / 8, 256, 0, stream>>>(router_input, w_router, logits,
                                           tlist, wlist, counts);

  const int grid = E * (T / MT);
  if (bf16_path) {
    // One-time (per launch) down-conversion: weights become 100 MB total,
    // fully L2-resident; hot loops are then pure b128 loads + v_wmma.
    cvt_bf16_kernel<<<2048, 256, 0, stream>>>(hidden, Xbf, nX);
    cvt_bf16_kernel<<<4096, 256, 0, stream>>>(w_gate, Wgbf, nW);
    cvt_bf16_kernel<<<4096, 256, 0, stream>>>(w_up, Wubf, nW);
    cvt_bf16_kernel<<<4096, 256, 0, stream>>>(w_down, Wdbf, nW);
    expert_kernel<__bf16, __bf16><<<grid, 256, 0, stream>>>(
        Xbf, Wgbf, Wubf, Wdbf, tlist, wlist, counts, out);
  } else {
    expert_kernel<float, float><<<grid, 256, 0, stream>>>(
        hidden, w_gate, w_up, w_down, tlist, wlist, counts, out);
  }
}
